// TGN_94489281061
// MI455X (gfx1250) — matrix operand order repaired and checked by reading, not compiled, not run on hardware
//
#include <hip/hip_runtime.h>
#include <hip/hip_bf16.h>
#include <cstdint>

// ---- problem constants (fixed by the reference) ----
#define M_DIM   128
#define FE_DIM  128
#define FN_DIM  128
#define C_DIM   10
#define MSG_H   256
#define EMB_HID 192

typedef __attribute__((ext_vector_type(16))) _Float16     v16h;
typedef __attribute__((ext_vector_type(8)))  float        v8f;
typedef __attribute__((ext_vector_type(4)))  float        f32x4;
typedef __attribute__((ext_vector_type(4)))  unsigned int u32x4;
typedef __attribute__((ext_vector_type(4)))  int          i32x4;

#if __has_builtin(__builtin_amdgcn_global_load_async_to_lds_b128) && \
    __has_builtin(__builtin_amdgcn_s_wait_asynccnt)
#define ASYNC_LDS 1
#else
#define ASYNC_LDS 0
#endif

// ---------- cooperative weight staging: global -> LDS ----------
__device__ __forceinline__ void stage_to_lds(const _Float16* __restrict__ g,
                                             _Float16* l, int nhalfs) {
    int tid = threadIdx.x, nth = blockDim.x;
#if ASYNC_LDS
    for (int i = tid * 8; i < nhalfs; i += nth * 8)
        __builtin_amdgcn_global_load_async_to_lds_b128(
            (__attribute__((address_space(1))) i32x4*)(g + i),
            (__attribute__((address_space(3))) i32x4*)(l + i), 0, 0);
    __builtin_amdgcn_s_wait_asynccnt(0);
#else
    for (int i = tid * 8; i < nhalfs; i += nth * 8)
        *(u32x4*)(l + i) = *(const u32x4*)(g + i);
#endif
    __syncthreads();
}

// ---------- fragment helpers ----------
// B block in LDS: 512 f16 = 32 lanes x 16, stored as [32 x lo-16B][32 x hi-16B]
// -> each ds_load_b128 has lanes 0..15 spanning all 64 banks exactly once.
__device__ __forceinline__ v16h load_b_lds(const _Float16* blk, int lane) {
    union { u32x4 v[2]; v16h h; } u;
    u.v[0] = *(const u32x4*)(blk + lane * 8);
    u.v[1] = *(const u32x4*)(blk + 256 + lane * 8);
    return u.h;
}

// A fragment from f32 row data (converted to f16).
__device__ __forceinline__ v16h a_from_f32(const float* p0, const float* p1) {
    f32x4 a0 = *(const f32x4*)(p0);
    f32x4 a1 = *(const f32x4*)(p0 + 4);
    f32x4 b0 = *(const f32x4*)(p1);
    f32x4 b1 = *(const f32x4*)(p1 + 4);
    v16h r;
#pragma unroll
    for (int j = 0; j < 4; ++j) {
        r[j]      = (_Float16)a0[j];
        r[4 + j]  = (_Float16)a1[j];
        r[8 + j]  = (_Float16)b0[j];
        r[12 + j] = (_Float16)b1[j];
    }
    return r;
}

// A fragment from f16 row data (global or LDS), 16B-aligned chunks.
__device__ __forceinline__ v16h a_from_f16(const _Float16* p0, const _Float16* p1) {
    union { u32x4 v[2]; v16h h; } u;
    u.v[0] = *(const u32x4*)p0;
    u.v[1] = *(const u32x4*)p1;
    return u.h;
}

__device__ __forceinline__ v8f wmma(v16h a, v16h b, v8f c) {
    return __builtin_amdgcn_wmma_f32_16x16x32_f16(false, a, false, b, (short)0, c,
                                                  false, false);
}

__device__ __forceinline__ float leaky(float x) { return fmaxf(x, 0.1f * x); }
__device__ __forceinline__ float sigm(float x)  { return 1.0f / (1.0f + __expf(-x)); }

// ---------- weight repack: W[Nout,Kin] f32 -> B-fragment f16 blocks ----------
// value(lane l, elem j) = W[nc*16 + (l&15)][kc*32 + (l<16?0:8) + (j<8 ? j : 8+j)]
// stored split lo/hi: j<8 at blk*512 + l*8 + j ; j>=8 at blk*512 + 256 + l*8 + j-8
__global__ void pack_w(const float* __restrict__ W, _Float16* __restrict__ P,
                       int Nout, int Kin, int NoutPad) {
    int id = blockIdx.x * blockDim.x + threadIdx.x;
    int KC = Kin >> 5;
    int total = (NoutPad >> 4) * KC * 512;
    if (id >= total) return;
    int j   = id & 15;
    int l   = (id >> 4) & 31;
    int blk = id >> 9;
    int nc = blk / KC, kc = blk % KC;
    int n = nc * 16 + (l & 15);
    int k = kc * 32 + ((l < 16) ? 0 : 8) + ((j < 8) ? j : 8 + j);
    float v = (n < Nout) ? W[(size_t)n * Kin + k] : 0.0f;
    int within = (j < 8) ? (l * 8 + j) : (256 + l * 8 + (j - 8));
    P[(size_t)blk * 512 + within] = (_Float16)v;
}

// ---------- scatter-collision resolution ----------
__global__ void init_win(int* __restrict__ win, int N) {
    int i = blockIdx.x * blockDim.x + threadIdx.x;
    if (i < N) win[i] = -1;
}
__global__ void scatter_win(const int* __restrict__ src, const int* __restrict__ dst,
                            int* __restrict__ win, int E) {
    int e = blockIdx.x * blockDim.x + threadIdx.x;
    if (e >= E) return;
    atomicMax(&win[src[e]], e);          // src pass: key = e
    atomicMax(&win[dst[e]], E + e);      // dst pass beats every src key
}

// ---------- kernel 1: msg MLP hidden  H = leaky([memA|memB|te|ef] @ W1^T + b1) ----------
// dynamic LDS: 16*14*512 halfs (224KB) of packed W1
__global__ __launch_bounds__(512, 1)
void msg_hidden_kernel(const float* __restrict__ memory,
                       const float* __restrict__ last_update,
                       const int* __restrict__ idxA, const int* __restrict__ idxB,
                       const float* __restrict__ event_t,
                       const float* __restrict__ edge_feat,
                       const float* __restrict__ w_time,
                       const float* __restrict__ b_time,
                       const _Float16* __restrict__ W1p, const float* __restrict__ b1,
                       _Float16* __restrict__ H, int E) {
    extern __shared__ _Float16 smem[];
    stage_to_lds(W1p, smem, 16 * 14 * 512);

    int lane = threadIdx.x & 31;
    int wpb  = blockDim.x >> 5;
    int wave0  = blockIdx.x * wpb + (threadIdx.x >> 5);
    int nwaves = gridDim.x * wpb;
    int ntiles = (E + 15) >> 4;
    int m    = lane & 15;
    int hi8  = (lane < 16) ? 0 : 8;
    int koff = (lane < 16) ? 0 : 8;

    for (int t = wave0; t < ntiles; t += nwaves) {
        bool full = (t * 16 + 16 <= E);
        int e = t * 16 + m; if (e >= E) e = E - 1;
        int ia = idxA[e], ib = idxB[e];
        float dt = event_t[e] - last_update[ia];
        const float* rowA = memory    + (size_t)ia * M_DIM;
        const float* rowB = memory    + (size_t)ib * M_DIM;
        const float* rowF = edge_feat + (size_t)e  * FE_DIM;

        v16h a[14];
#pragma unroll
        for (int c = 0; c < 4; ++c)
            a[c] = a_from_f32(rowA + c * 32 + koff, rowA + c * 32 + 16 + koff);
#pragma unroll
        for (int c = 0; c < 4; ++c)
            a[4 + c] = a_from_f32(rowB + c * 32 + koff, rowB + c * 32 + 16 + koff);
#pragma unroll
        for (int c = 0; c < 2; ++c) {   // time encode: cos(dt*w + b)
            v16h tv;
#pragma unroll
            for (int j = 0; j < 8; ++j) {
                int t0 = c * 32 + koff + j;
                int t1 = c * 32 + 16 + koff + j;
                tv[j]     = (_Float16)__cosf(dt * w_time[t0] + b_time[t0]);
                tv[8 + j] = (_Float16)__cosf(dt * w_time[t1] + b_time[t1]);
            }
            a[8 + c] = tv;
        }
#pragma unroll
        for (int c = 0; c < 4; ++c)
            a[10 + c] = a_from_f32(rowF + c * 32 + koff, rowF + c * 32 + 16 + koff);

        for (int nc = 0; nc < 16; ++nc) {           // N = 256
            int col = nc * 16 + m;
            float bias = b1[col];                   // issued before WMMA chain
            const _Float16* wb = smem + (size_t)(nc * 14) * 512;
            v8f acc = {};
            v16h b = load_b_lds(wb, lane);          // double-buffered B
#pragma unroll
            for (int c = 0; c < 13; ++c) {
                v16h bn = load_b_lds(wb + (c + 1) * 512, lane);
                acc = wmma(a[c], b, acc);
                b = bn;
            }
            acc = wmma(a[13], b, acc);

            _Float16* hcol = H + (size_t)(t * 16) * MSG_H + col;
            if (full) {
#pragma unroll
                for (int r = 0; r < 8; ++r)
                    hcol[(size_t)(r + hi8) * MSG_H] = (_Float16)leaky(acc[r] + bias);
            } else {
#pragma unroll
                for (int r = 0; r < 8; ++r) {
                    int e2 = t * 16 + r + hi8;
                    if (e2 < E)
                        hcol[(size_t)(r + hi8) * MSG_H] = (_Float16)leaky(acc[r] + bias);
                }
            }
        }
    }
}

// ---------- kernel 2: msg layer2 + GRU  up = GRU(H@W2^T+b2, memory[idx]) ----------
// dynamic LDS: W2 (32768) + Wih (49152) + Whh (49152) + 8 transpose tiles (2176 ea)
#define K2_W2   0
#define K2_WIH  32768
#define K2_WHH  81920
#define K2_TILE 131072
__global__ __launch_bounds__(256, 1)
void msg_gru_kernel(const _Float16* __restrict__ H,
                    const _Float16* __restrict__ W2p, const float* __restrict__ b2,
                    const _Float16* __restrict__ Wihp, const _Float16* __restrict__ Whhp,
                    const float* __restrict__ bih, const float* __restrict__ bhh,
                    const float* __restrict__ memory, const int* __restrict__ idx,
                    float* __restrict__ up, int E) {
    extern __shared__ _Float16 smem[];
    stage_to_lds(W2p,  smem + K2_W2,  32768);
    stage_to_lds(Wihp, smem + K2_WIH, 49152);
    stage_to_lds(Whhp, smem + K2_WHH, 49152);

    int lane = threadIdx.x & 31;
    int wib  = threadIdx.x >> 5;
    int wpb  = blockDim.x >> 5;
    int wave0  = blockIdx.x * wpb + wib;
    int nwaves = gridDim.x * wpb;
    int ntiles = (E + 15) >> 4;
    int m    = lane & 15;
    int hi8  = (lane < 16) ? 0 : 8;
    int koff = (lane < 16) ? 0 : 8;
    _Float16* tile = smem + K2_TILE + wib * 2176;   // 16 x 128, stride 136

    for (int t = wave0; t < ntiles; t += nwaves) {
        bool full = (t * 16 + 16 <= E);
        int e = t * 16 + m; if (e >= E) e = E - 1;
        const _Float16* hrow = H + (size_t)e * MSG_H;
        v16h aH[8];
#pragma unroll
        for (int c = 0; c < 8; ++c)
            aH[c] = a_from_f16(hrow + c * 32 + koff, hrow + c * 32 + 16 + koff);

        for (int nc = 0; nc < 8; ++nc) {            // msg = H @ W2^T + b2
            int col = nc * 16 + m;
            float bias = b2[col];
            const _Float16* wb = smem + K2_W2 + (size_t)(nc * 8) * 512;
            v8f acc = {};
            v16h b = load_b_lds(wb, lane);
#pragma unroll
            for (int c = 0; c < 7; ++c) {
                v16h bn = load_b_lds(wb + (c + 1) * 512, lane);
                acc = wmma(aH[c], b, acc);
                b = bn;
            }
            acc = wmma(aH[7], b, acc);
#pragma unroll
            for (int r = 0; r < 8; ++r)             // C-layout -> LDS transpose
                tile[(r + hi8) * 136 + col] = (_Float16)(acc[r] + bias);
        }

        v16h aM[4], aX[4];
#pragma unroll
        for (int c = 0; c < 4; ++c)
            aM[c] = a_from_f16(tile + m * 136 + c * 32 + koff,
                               tile + m * 136 + c * 32 + 16 + koff);
        const float* xrow = memory + (size_t)idx[e] * M_DIM;
#pragma unroll
        for (int c = 0; c < 4; ++c)
            aX[c] = a_from_f32(xrow + c * 32 + koff, xrow + c * 32 + 16 + koff);

        int irow[8];                                // gather rows (nc-invariant)
#pragma unroll
        for (int r = 0; r < 8; ++r) {
            int e2 = t * 16 + r + hi8;
            irow[r] = idx[(e2 < E) ? e2 : (E - 1)];
        }

        for (int nc = 0; nc < 8; ++nc) {            // gates r,z,n
            int col = nc * 16 + m;
            float b_ir = bih[col],       b_hr = bhh[col];
            float b_iz = bih[128 + col], b_hz = bhh[128 + col];
            float b_in = bih[256 + col], b_hn = bhh[256 + col];
            float hv[8];
#pragma unroll
            for (int r = 0; r < 8; ++r)             // old memory values (in flight)
                hv[r] = memory[(size_t)irow[r] * M_DIM + col];

            v8f air = {}, aiz = {}, ain = {}, ahr = {}, ahz = {}, ahn = {};
            const _Float16* wi = smem + K2_WIH;
            const _Float16* wh = smem + K2_WHH;
#pragma unroll
            for (int c = 0; c < 4; ++c) {           // batch 6 B loads per 6 WMMA
                v16h bir = load_b_lds(wi + (size_t)((nc)      * 4 + c) * 512, lane);
                v16h biz = load_b_lds(wi + (size_t)((nc + 8)  * 4 + c) * 512, lane);
                v16h bin = load_b_lds(wi + (size_t)((nc + 16) * 4 + c) * 512, lane);
                v16h bhr = load_b_lds(wh + (size_t)((nc)      * 4 + c) * 512, lane);
                v16h bhz = load_b_lds(wh + (size_t)((nc + 8)  * 4 + c) * 512, lane);
                v16h bhn = load_b_lds(wh + (size_t)((nc + 16) * 4 + c) * 512, lane);
                air = wmma(aM[c], bir, air);
                aiz = wmma(aM[c], biz, aiz);
                ain = wmma(aM[c], bin, ain);
                ahr = wmma(aX[c], bhr, ahr);
                ahz = wmma(aX[c], bhz, ahz);
                ahn = wmma(aX[c], bhn, ahn);
            }
            float* ucol = up + (size_t)(t * 16) * M_DIM + col;
#pragma unroll
            for (int r = 0; r < 8; ++r) {
                float rg = sigm((air[r] + b_ir) + (ahr[r] + b_hr));
                float zg = sigm((aiz[r] + b_iz) + (ahz[r] + b_hz));
                float ng = tanhf((ain[r] + b_in) + rg * (ahn[r] + b_hn));
                float u  = (1.0f - zg) * ng + zg * hv[r];
                if (full) {
                    ucol[(size_t)(r + hi8) * M_DIM] = u;
                } else {
                    int e2 = t * 16 + r + hi8;
                    if (e2 < E) ucol[(size_t)(r + hi8) * M_DIM] = u;
                }
            }
        }
    }
}

// ---------- kernel 4: emb hidden  G = leaky([new_mem[src]|feat[src]] @ We1^T + be1) ----------
// dynamic LDS: 12*8*512 halfs (96KB)
__global__ __launch_bounds__(512, 1)
void emb_hidden_kernel(const float* __restrict__ memory,
                       const float* __restrict__ node_feat,
                       const int* __restrict__ src, const int* __restrict__ win,
                       const float* __restrict__ up_s, const float* __restrict__ up_d,
                       const _Float16* __restrict__ We1p, const float* __restrict__ be1,
                       _Float16* __restrict__ G, int E) {
    extern __shared__ _Float16 smem[];
    stage_to_lds(We1p, smem, 12 * 8 * 512);

    int lane = threadIdx.x & 31;
    int wpb  = blockDim.x >> 5;
    int wave0  = blockIdx.x * wpb + (threadIdx.x >> 5);
    int nwaves = gridDim.x * wpb;
    int ntiles = (E + 15) >> 4;
    int m    = lane & 15;
    int hi8  = (lane < 16) ? 0 : 8;
    int koff = (lane < 16) ? 0 : 8;

    for (int t = wave0; t < ntiles; t += nwaves) {
        bool full = (t * 16 + 16 <= E);
        int e = t * 16 + m; if (e >= E) e = E - 1;
        int n = src[e];
        int k = win[n];                          // winner: dst-pass beats src-pass
        const float* mrow = (k < 0) ? (memory + (size_t)n * M_DIM)
                          : (k < E) ? (up_s + (size_t)k * M_DIM)
                                    : (up_d + (size_t)(k - E) * M_DIM);
        const float* frow = node_feat + (size_t)n * FN_DIM;

        v16h a[8];
#pragma unroll
        for (int c = 0; c < 4; ++c)
            a[c] = a_from_f32(mrow + c * 32 + koff, mrow + c * 32 + 16 + koff);
#pragma unroll
        for (int c = 0; c < 4; ++c)
            a[4 + c] = a_from_f32(frow + c * 32 + koff, frow + c * 32 + 16 + koff);

        for (int nc = 0; nc < 12; ++nc) {           // N = 192
            int col = nc * 16 + m;
            float bias = be1[col];
            const _Float16* wb = smem + (size_t)(nc * 8) * 512;
            v8f acc = {};
            v16h b = load_b_lds(wb, lane);
#pragma unroll
            for (int c = 0; c < 7; ++c) {
                v16h bn = load_b_lds(wb + (c + 1) * 512, lane);
                acc = wmma(a[c], b, acc);
                b = bn;
            }
            acc = wmma(a[7], b, acc);

            _Float16* gcol = G + (size_t)(t * 16) * EMB_HID + col;
            if (full) {
#pragma unroll
                for (int r = 0; r < 8; ++r)
                    gcol[(size_t)(r + hi8) * EMB_HID] = (_Float16)leaky(acc[r] + bias);
            } else {
#pragma unroll
                for (int r = 0; r < 8; ++r) {
                    int e2 = t * 16 + r + hi8;
                    if (e2 < E)
                        gcol[(size_t)(r + hi8) * EMB_HID] = (_Float16)leaky(acc[r] + bias);
                }
            }
        }
    }
}

// ---------- kernel 5: emb layer2 + classifier ----------
// dynamic LDS: We2 (24576) + Wc (2048) + 8 transpose tiles (2176 ea)
#define K5_WE2  0
#define K5_WC   24576
#define K5_TILE 26624
__global__ __launch_bounds__(256, 1)
void emb_out_kernel(const _Float16* __restrict__ G,
                    const _Float16* __restrict__ We2p, const float* __restrict__ be2,
                    const _Float16* __restrict__ Wcp,  const float* __restrict__ bc,
                    float* __restrict__ out, int E) {
    extern __shared__ _Float16 smem[];
    stage_to_lds(We2p, smem + K5_WE2, 24576);
    stage_to_lds(Wcp,  smem + K5_WC,  2048);

    int lane = threadIdx.x & 31;
    int wib  = threadIdx.x >> 5;
    int wpb  = blockDim.x >> 5;
    int wave0  = blockIdx.x * wpb + wib;
    int nwaves = gridDim.x * wpb;
    int ntiles = (E + 15) >> 4;
    int m    = lane & 15;
    int hi8  = (lane < 16) ? 0 : 8;
    int koff = (lane < 16) ? 0 : 8;
    _Float16* tile = smem + K5_TILE + wib * 2176;
    float bias_c = (m < C_DIM) ? bc[m] : 0.0f;

    for (int t = wave0; t < ntiles; t += nwaves) {
        bool full = (t * 16 + 16 <= E);
        int e = t * 16 + m; if (e >= E) e = E - 1;
        const _Float16* grow = G + (size_t)e * EMB_HID;
        v16h aG[6];
#pragma unroll
        for (int c = 0; c < 6; ++c)                 // K = 192
            aG[c] = a_from_f16(grow + c * 32 + koff, grow + c * 32 + 16 + koff);

        for (int nc = 0; nc < 8; ++nc) {            // emb = G @ We2^T + be2
            int col = nc * 16 + m;
            float bias = be2[col];
            const _Float16* wb = smem + K5_WE2 + (size_t)(nc * 6) * 512;
            v8f acc = {};
            v16h b = load_b_lds(wb, lane);
#pragma unroll
            for (int c = 0; c < 5; ++c) {
                v16h bn = load_b_lds(wb + (c + 1) * 512, lane);
                acc = wmma(aG[c], b, acc);
                b = bn;
            }
            acc = wmma(aG[5], b, acc);
#pragma unroll
            for (int r = 0; r < 8; ++r)
                tile[(r + hi8) * 136 + col] = (_Float16)(acc[r] + bias);
        }

        v16h aE[4];
#pragma unroll
        for (int c = 0; c < 4; ++c)
            aE[c] = a_from_f16(tile + m * 136 + c * 32 + koff,
                               tile + m * 136 + c * 32 + 16 + koff);
        v8f acc = {};                               // logits vs Wc padded 10->16
        {
            v16h b = load_b_lds(smem + K5_WC, lane);
#pragma unroll
            for (int c = 0; c < 3; ++c) {
                v16h bn = load_b_lds(smem + K5_WC + (size_t)(c + 1) * 512, lane);
                acc = wmma(aE[c], b, acc);
                b = bn;
            }
            acc = wmma(aE[3], b, acc);
        }
        float* ocol = out + (size_t)(t * 16) * C_DIM + m;
        if (m < C_DIM) {
            if (full) {
#pragma unroll
                for (int r = 0; r < 8; ++r)
                    ocol[(size_t)(r + hi8) * C_DIM] = acc[r] + bias_c;
            } else {
#pragma unroll
                for (int r = 0; r < 8; ++r) {
                    int e2 = t * 16 + r + hi8;
                    if (e2 < E) ocol[(size_t)(r + hi8) * C_DIM] = acc[r] + bias_c;
                }
            }
        }
    }
}

// ---------------- host-side launch ----------------
extern "C" void kernel_launch(void* const* d_in, const int* in_sizes, int n_in,
                              void* d_out, int out_size, void* d_ws, size_t ws_size,
                              hipStream_t stream) {
    const float* memory      = (const float*)d_in[0];
    const float* last_update = (const float*)d_in[1];
    const int*   src         = (const int*)d_in[2];
    const int*   dst         = (const int*)d_in[3];
    const float* event_t     = (const float*)d_in[4];
    const float* edge_feat   = (const float*)d_in[5];
    const float* node_feat   = (const float*)d_in[6];
    const float* w_time      = (const float*)d_in[7];
    const float* b_time      = (const float*)d_in[8];
    const float* Ws1 = (const float*)d_in[9];  const float* bs1 = (const float*)d_in[10];
    const float* Ws2 = (const float*)d_in[11]; const float* bs2 = (const float*)d_in[12];
    const float* Wd1 = (const float*)d_in[13]; const float* bd1 = (const float*)d_in[14];
    const float* Wd2 = (const float*)d_in[15]; const float* bd2 = (const float*)d_in[16];
    const float* Wih = (const float*)d_in[17]; const float* Whh = (const float*)d_in[18];
    const float* bih = (const float*)d_in[19]; const float* bhh = (const float*)d_in[20];
    const float* We1 = (const float*)d_in[21]; const float* be1 = (const float*)d_in[22];
    const float* We2 = (const float*)d_in[23]; const float* be2 = (const float*)d_in[24];
    const float* Wc  = (const float*)d_in[25]; const float* bc  = (const float*)d_in[26];
    float* out = (float*)d_out;

    const int N = in_sizes[0] / M_DIM;
    const int E = in_sizes[2];

    // ---- workspace carve-out (256B aligned) ----
    char* ws = (char*)d_ws;
    size_t off = 0;
    auto take = [&](size_t bytes) -> void* {
        void* p = (void*)(ws + off);
        off += (bytes + 255) & ~(size_t)255;
        return p;
    };
    int*      win  = (int*)take((size_t)N * 4);
    _Float16* Ws1p = (_Float16*)take((size_t)256 * 448 * 2);
    _Float16* Wd1p = (_Float16*)take((size_t)256 * 448 * 2);
    _Float16* Ws2p = (_Float16*)take((size_t)128 * 256 * 2);
    _Float16* Wd2p = (_Float16*)take((size_t)128 * 256 * 2);
    _Float16* Wihp = (_Float16*)take((size_t)384 * 128 * 2);
    _Float16* Whhp = (_Float16*)take((size_t)384 * 128 * 2);
    _Float16* We1p = (_Float16*)take((size_t)192 * 256 * 2);
    _Float16* We2p = (_Float16*)take((size_t)128 * 192 * 2);
    _Float16* Wcp  = (_Float16*)take((size_t)16  * 128 * 2);
    float*    up_s = (float*)take((size_t)E * M_DIM * 4);
    float*    up_d = (float*)take((size_t)E * M_DIM * 4);
    _Float16* H    = (_Float16*)take((size_t)E * MSG_H * 2);
    _Float16* G    = H;                       // H dead before G written

    // dynamic-LDS sizes
    const int SH1 = 16 * 14 * 512 * 2;                    // 229376 B
    const int SH2 = (131072 + 8 * 2176) * 2;              // 296960 B
    const int SH4 = 12 * 8 * 512 * 2;                     //  98304 B
    const int SH5 = (26624 + 8 * 2176) * 2;               //  88064 B
    (void)hipFuncSetAttribute((const void*)msg_hidden_kernel,
                              hipFuncAttributeMaxDynamicSharedMemorySize, SH1);
    (void)hipFuncSetAttribute((const void*)msg_gru_kernel,
                              hipFuncAttributeMaxDynamicSharedMemorySize, SH2);
    (void)hipFuncSetAttribute((const void*)emb_hidden_kernel,
                              hipFuncAttributeMaxDynamicSharedMemorySize, SH4);
    (void)hipFuncSetAttribute((const void*)emb_out_kernel,
                              hipFuncAttributeMaxDynamicSharedMemorySize, SH5);

    auto pack = [&](const float* W, _Float16* P, int Nout, int Kin, int NoutPad) {
        int total = (NoutPad / 16) * (Kin / 32) * 512;
        pack_w<<<dim3((total + 255) / 256), dim3(256), 0, stream>>>(W, P, Nout, Kin, NoutPad);
    };
    pack(Ws1, Ws1p, 256, 448, 256);
    pack(Wd1, Wd1p, 256, 448, 256);
    pack(Ws2, Ws2p, 128, 256, 128);
    pack(Wd2, Wd2p, 128, 256, 128);
    pack(Wih, Wihp, 384, 128, 384);
    pack(Whh, Whhp, 384, 128, 384);
    pack(We1, We1p, 192, 256, 192);
    pack(We2, We2p, 128, 192, 128);
    pack(Wc,  Wcp,  10,  128, 16);

    init_win<<<dim3((N + 255) / 256), dim3(256), 0, stream>>>(win, N);
    scatter_win<<<dim3((E + 255) / 256), dim3(256), 0, stream>>>(src, dst, win, E);

    int ntiles = (E + 15) / 16;
    auto blocksFor = [&](int want, int wpb) {
        int need = (ntiles + wpb - 1) / wpb;
        return need < want ? need : want;
    };
    dim3 g1(blocksFor(512, 16)), b1(512);
    dim3 g2(blocksFor(512, 8)),  b2v(256);

    // src side
    msg_hidden_kernel<<<g1, b1, SH1, stream>>>(memory, last_update, src, dst,
        event_t, edge_feat, w_time, b_time, Ws1p, bs1, H, E);
    msg_gru_kernel<<<g2, b2v, SH2, stream>>>(H, Ws2p, bs2, Wihp, Whhp, bih, bhh,
        memory, src, up_s, E);
    // dst side
    msg_hidden_kernel<<<g1, b1, SH1, stream>>>(memory, last_update, dst, src,
        event_t, edge_feat, w_time, b_time, Wd1p, bd1, H, E);
    msg_gru_kernel<<<g2, b2v, SH2, stream>>>(H, Wd2p, bd2, Wihp, Whhp, bih, bhh,
        memory, dst, up_d, E);

    emb_hidden_kernel<<<g1, b1, SH4, stream>>>(memory, node_feat, src, win,
        up_s, up_d, We1p, be1, G, E);
    emb_out_kernel<<<g2, b2v, SH5, stream>>>(G, We2p, be2, Wcp, bc, out, E);
}
